// FaceEuclideanWrapper_35828617183575
// MI455X (gfx1250) — matrix operand order repaired and checked by reading, not compiled, not run on hardware
//
#include <hip/hip_runtime.h>
#include <hip/hip_bf16.h>

typedef __attribute__((ext_vector_type(16))) _Float16 v16h;
typedef __attribute__((ext_vector_type(8)))  _Float16 v8h;
typedef __attribute__((ext_vector_type(8)))  float    v8f;

#define B_ROWS 2048
#define DIM    512
#define N_DB   50000
#define HW     49
#define NSPLIT 25
#define STRIP  2000          // N_DB / NSPLIT
#define STRIP_TILES 125      // STRIP / 16

// ---------------- 1) global average pool + ||a||^2 ----------------
__global__ void gap_kernel(const float* __restrict__ x,
                           _Float16* __restrict__ emb,
                           float* __restrict__ a2) {
    __shared__ float red[256];
    int b = blockIdx.x;
    float part = 0.f;
    for (int c = threadIdx.x; c < DIM; c += 256) {
        const float* p = x + ((size_t)b * DIM + c) * HW;
        float s = 0.f;
        #pragma unroll
        for (int i = 0; i < HW; ++i) s += p[i];
        float avg = s * (1.0f / 49.0f);
        emb[(size_t)b * DIM + c] = (_Float16)avg;
        part += avg * avg;
    }
    red[threadIdx.x] = part;
    __syncthreads();
    for (int off = 128; off > 0; off >>= 1) {
        if (threadIdx.x < off) red[threadIdx.x] += red[threadIdx.x + off];
        __syncthreads();
    }
    if (threadIdx.x == 0) a2[b] = red[0];
}

// ---------------- 2) db -> f16 + ||b||^2 (one wave32 per row) ----------------
__global__ void dbprep_kernel(const float* __restrict__ db,
                              _Float16* __restrict__ dbh,
                              float* __restrict__ b2) {
    int wave = threadIdx.x >> 5, lane = threadIdx.x & 31;
    int n = blockIdx.x * 8 + wave;
    const float* src = db + (size_t)n * DIM + lane * 16;
    _Float16*   dst  = dbh + (size_t)n * DIM + lane * 16;
    float ss = 0.f;
    #pragma unroll
    for (int i = 0; i < 16; ++i) {
        float v = src[i];
        ss += v * v;
        dst[i] = (_Float16)v;
    }
    #pragma unroll
    for (int m = 16; m > 0; m >>= 1) ss += __shfl_xor(ss, m, 32);
    if (lane == 0) b2[n] = ss;
}

// ---------------- 3) init packed (d2,idx) accumulators ----------------
__global__ void init_kernel(unsigned long long* __restrict__ out64) {
    int i = blockIdx.x * 256 + threadIdx.x;
    if (i < B_ROWS) out64[i] = ~0ull;
}

// ---------------- 4) WMMA GEMM fused with min/argmin epilogue ----------------
// Block = 2 waves; each wave owns 32 rows (two 16x16 accumulators sharing one
// B fragment -> halves B bytes per WMMA and interleaves two independent
// accumulation chains to hide WMMA->WMMA latency).
__global__ __launch_bounds__(64) void gemm_min_kernel(
    const _Float16* __restrict__ embh, const _Float16* __restrict__ dbh,
    const float* __restrict__ a2, const float* __restrict__ b2,
    unsigned long long* __restrict__ out64) {

    __shared__ _Float16 As[64 * DIM];   // 64 KB A tile (64 rows x 512)
    int tid  = threadIdx.x;
    int wave = tid >> 5;      // 0..1
    int lane = tid & 31;
    int hi   = lane >> 4;     // 0 or 1 (half-wave)
    int ln   = lane & 15;

    int mBlock = blockIdx.x * 64;
    // cooperative A-tile stage: 4096 uint4 over 64 threads
    {
        const uint4* src = (const uint4*)(embh + (size_t)mBlock * DIM);
        uint4* dst = (uint4*)As;
        for (int i = tid; i < 64 * DIM / 8; i += 64) dst[i] = src[i];
    }
    __syncthreads();

    int mBase = mBlock + wave * 32;                 // 32 rows per wave
    const _Float16* Arow0 = As + (size_t)(wave * 32 + ln) * DIM;  // rows m=ln
    const _Float16* Arow1 = Arow0 + (size_t)16 * DIM;             // rows m=16+ln

    float a2v[16];
    #pragma unroll
    for (int s = 0; s < 2; ++s)
        #pragma unroll
        for (int j = 0; j < 8; ++j)
            a2v[s * 8 + j] = a2[mBase + s * 16 + j + hi * 8];

    float    runD[16];
    unsigned runI[16];
    #pragma unroll
    for (int j = 0; j < 16; ++j) { runD[j] = 3.402823466e38f; runI[j] = 0u; }

    int n0base = blockIdx.y * STRIP;

    for (int t = 0; t < STRIP_TILES; ++t) {
        int n0 = n0base + t * 16;
        int nB = n0 + ln;                           // B: column n = lane&15
        const _Float16* bptr = dbh + (size_t)nB * DIM + hi * 16;
        if (t + 1 < STRIP_TILES)                    // global_prefetch_b8 next tile
            __builtin_prefetch(bptr + 16 * DIM, 0, 1);
        v8f c0 = {};
        v8f c1 = {};
        #pragma unroll 4
        for (int kk = 0; kk < 16; ++kk) {
            int k0 = kk * 32 + hi * 16;
            v16h bf = *(const v16h*)(bptr + kk * 32);  // shared by both subtiles
            v8h a0l = *(const v8h*)(Arow0 + k0);
            v8h a0h = *(const v8h*)(Arow0 + k0 + 8);
            v16h af0 = __builtin_shufflevector(a0l, a0h,
                        0,1,2,3,4,5,6,7,8,9,10,11,12,13,14,15);
            v8h a1l = *(const v8h*)(Arow1 + k0);
            v8h a1h = *(const v8h*)(Arow1 + k0 + 8);
            v16h af1 = __builtin_shufflevector(a1l, a1h,
                        0,1,2,3,4,5,6,7,8,9,10,11,12,13,14,15);
            c0 = __builtin_amdgcn_wmma_f32_16x16x32_f16(
                    false, af0, false, bf, (short)0, c0, false, false);
            c1 = __builtin_amdgcn_wmma_f32_16x16x32_f16(
                    false, af1, false, bf, (short)0, c1, false, false);
        }
        // min over clamped d^2 (sqrt is monotonic -> defer it to finalize)
        float b2n = b2[n0 + ln];
        unsigned nIdx = (unsigned)(n0 + ln);
        #pragma unroll
        for (int j = 0; j < 8; ++j) {
            float d2 = fmaxf(fmaf(-2.0f, c0[j], a2v[j] + b2n), 0.0f);
            bool lt = d2 < runD[j];        // strict < keeps first (smallest n)
            runD[j] = lt ? d2   : runD[j];
            runI[j] = lt ? nIdx : runI[j];
        }
        #pragma unroll
        for (int j = 0; j < 8; ++j) {
            float d2 = fmaxf(fmaf(-2.0f, c1[j], a2v[8 + j] + b2n), 0.0f);
            bool lt = d2 < runD[8 + j];
            runD[8 + j] = lt ? d2   : runD[8 + j];
            runI[8 + j] = lt ? nIdx : runI[8 + j];
        }
    }

    // min over the 16 lanes that share a row (each half-wave independently),
    // lexicographic (d2, idx) to keep argmin first-occurrence semantics
    #pragma unroll
    for (int s = 0; s < 2; ++s) {
        #pragma unroll
        for (int j = 0; j < 8; ++j) {
            float d = runD[s * 8 + j];
            unsigned i = runI[s * 8 + j];
            #pragma unroll
            for (int m = 8; m > 0; m >>= 1) {
                float    od = __shfl_xor(d, m, 32);
                unsigned oi = __shfl_xor(i, m, 32);
                if (od < d || (od == d && oi < i)) { d = od; i = oi; }
            }
            if (ln == 0) {
                unsigned long long key =
                    ((unsigned long long)__float_as_uint(d) << 32) | i;
                atomicMin(&out64[mBase + s * 16 + j + hi * 8], key);
            }
        }
    }
}

// ---------------- 5) unpack: sqrt deferred to here ----------------
__global__ void finalize_kernel(const unsigned long long* __restrict__ out64,
                                float* __restrict__ outd, int* __restrict__ outi) {
    int b = blockIdx.x * 256 + threadIdx.x;
    if (b < B_ROWS) {
        unsigned long long k = out64[b];
        float d2 = __uint_as_float((unsigned)(k >> 32));
        outd[b] = sqrtf(d2);
        outi[b] = (int)(unsigned)(k & 0xffffffffu);
    }
}

extern "C" void kernel_launch(void* const* d_in, const int* in_sizes, int n_in,
                              void* d_out, int out_size, void* d_ws, size_t ws_size,
                              hipStream_t stream) {
    const float* x  = (const float*)d_in[0];
    const float* db = (const float*)d_in[1];

    char* ws = (char*)d_ws;
    size_t off = 0;
    _Float16* embh = (_Float16*)(ws + off); off += (size_t)B_ROWS * DIM * 2;  // 2 MB
    _Float16* dbh  = (_Float16*)(ws + off); off += (size_t)N_DB  * DIM * 2;   // 51.2 MB
    float* a2 = (float*)(ws + off); off += (size_t)B_ROWS * 4;
    float* b2 = (float*)(ws + off); off += (size_t)N_DB * 4;
    off = (off + 15) & ~(size_t)15;
    unsigned long long* out64 = (unsigned long long*)(ws + off);

    gap_kernel<<<B_ROWS, 256, 0, stream>>>(x, embh, a2);
    dbprep_kernel<<<N_DB / 8, 256, 0, stream>>>(db, dbh, b2);
    init_kernel<<<(B_ROWS + 255) / 256, 256, 0, stream>>>(out64);
    dim3 g(B_ROWS / 64, NSPLIT);
    gemm_min_kernel<<<g, 64, 0, stream>>>(embh, dbh, a2, b2, out64);
    finalize_kernel<<<(B_ROWS + 255) / 256, 256, 0, stream>>>(
        out64, (float*)d_out, (int*)d_out + B_ROWS);
}
